// VICRegLLoss_42717744726449
// MI455X (gfx1250) — matrix operand order repaired and checked by reading, not compiled, not run on hardware
//
#include <hip/hip_runtime.h>
#include <hip/hip_bf16.h>

// ---------------------------------------------------------------------------
// VICRegL loss for MI455X (gfx1250): bf16 WMMA for all GEMM-shaped work,
// f32 accumulation. Gram/Frobenius tricks avoid materializing DxD covariance;
// each wave register-blocks a 32x32 output tile (2x2 WMMAs) so every
// 128-bit fragment load feeds two v_wmma_f32_16x16x32_bf16 ops.
// ---------------------------------------------------------------------------

typedef __attribute__((ext_vector_type(16))) __bf16 v16bf;
typedef __attribute__((ext_vector_type(8)))  __bf16 v8bf;
typedef __attribute__((ext_vector_type(8)))  float  v8f;

#define GB 256      // global batch
#define GD 8192     // global dim
#define LB 256      // local batch
#define LN 256      // H*W
#define LC 1024     // local channels
#define NM 20       // num matches
#define LBM (LB*NM) // 5120 matched rows

#define LAMBDA_ 25.0f
#define MU_     25.0f
#define NU_     1.0f
#define ALPHA_  0.75f
#define EPS_    1e-4f

// scalar accumulator slots (in d_ws, zeroed each launch)
#define S_INV_G  0
#define S_VAR_G0 1
#define S_VAR_G1 2
#define S_DSQ_G0 3
#define S_DSQ_G1 4
#define S_FRO_G0 5
#define S_FRO_G1 6
#define S_PAIR(p) (8 + (p)*8)   // [inv, varX, varY, dsqX, dsqY, froX, froY]

// ---------------------------------------------------------------------------
// WMMA fragment loaders (bf16 16x16x32), vectorized.
// A (16x32): lanes 0-15 hold M=0..15 with K=0..15; lanes 16-31 hold
//            K=16..31  -> one contiguous 32B load per lane.
// B (32x16): lane n<16 holds col n K=0..15, lane n+16 K=16..31; loaded from
//            the *transposed* operand (rows of the second matrix) -> one
//            contiguous 32B load per lane.
// ---------------------------------------------------------------------------
__device__ inline v16bf frag_a_bf(const __bf16* p, size_t ld, int lane) {
  int m = lane & 15, h = lane >> 4;
  const __bf16* r = p + (size_t)m * ld + h * 16;
  return *reinterpret_cast<const v16bf*>(r);
}

__device__ inline v16bf frag_b_bf(const __bf16* p, size_t ld, int lane) {
  int n = lane & 15, h = lane >> 4;
  return *reinterpret_cast<const v16bf*>(p + (size_t)n * ld + h * 16);
}

__device__ inline v8f wmma_bf16(v16bf a, v16bf b, v8f c) {
  return __builtin_amdgcn_wmma_f32_16x16x32_bf16(false, a, false, b,
                                                 (short)0, c, false, false);
}

__device__ inline float lane_sum32(float v) {
#pragma unroll
  for (int m = 1; m < 32; m <<= 1) v += __shfl_xor(v, m);
  return v;
}

// 32x32 register-blocked tile of X·Xᵀ at (i0,j0): 2x2 WMMAs, fragments
// reused twice; returns the scaled Frobenius contribution of the block.
__device__ inline float gram_tile_fro32(const __bf16* x, size_t ld, int i0,
                                        int j0, int K, int lane, float scale) {
  v8f a00 = {}, a01 = {}, a10 = {}, a11 = {};
  for (int k = 0; k < K; k += 32) {
    __builtin_prefetch(x + (size_t)i0 * ld + k + 512, 0, 0);
    v16bf fa0 = frag_a_bf(x + (size_t)i0 * ld + k,        ld, lane);
    v16bf fa1 = frag_a_bf(x + (size_t)(i0+16) * ld + k,   ld, lane);
    v16bf fb0 = frag_b_bf(x + (size_t)j0 * ld + k,        ld, lane);
    v16bf fb1 = frag_b_bf(x + (size_t)(j0+16) * ld + k,   ld, lane);
    a00 = wmma_bf16(fa0, fb0, a00);
    a01 = wmma_bf16(fa0, fb1, a01);
    a10 = wmma_bf16(fa1, fb0, a10);
    a11 = wmma_bf16(fa1, fb1, a11);
  }
  float fro = 0.f;
#pragma unroll
  for (int v = 0; v < 8; ++v) {
    float c0 = a00[v] * scale, c1 = a01[v] * scale;
    float c2 = a10[v] * scale, c3 = a11[v] * scale;
    fro += c0*c0 + c1*c1 + c2*c2 + c3*c3;
  }
  return lane_sum32(fro);
}

// ---------------------------------------------------------------------------
__global__ void init_scalars_k(float* s) {
  if (threadIdx.x < 64) s[threadIdx.x] = 0.f;
}

// per-column stats of the two global views: means, var/diag terms, inv term
__global__ void global_colstats_k(const float* g0, const float* g1,
                                  float* mean0, float* mean1, float* s) {
  int d = blockIdx.x * blockDim.x + threadIdx.x;
  float s0 = 0, s1 = 0, q0 = 0, q1 = 0, inv = 0;
  for (int b = 0; b < GB; ++b) {
    float a = g0[(size_t)b*GD + d], c = g1[(size_t)b*GD + d];
    s0 += a; q0 += a*a; s1 += c; q1 += c*c;
    float df = a - c; inv += df*df;
  }
  float m0 = s0 / GB, m1 = s1 / GB;
  mean0[d] = m0; mean1[d] = m1;
  float v0 = (q0 - GB*m0*m0) / (GB - 1);
  float v1 = (q1 - GB*m1*m1) / (GB - 1);
  atomicAdd(&s[S_INV_G],  inv);
  atomicAdd(&s[S_VAR_G0], fmaxf(0.f, 1.f - sqrtf(v0 + EPS_)));
  atomicAdd(&s[S_VAR_G1], fmaxf(0.f, 1.f - sqrtf(v1 + EPS_)));
  atomicAdd(&s[S_DSQ_G0], v0*v0);
  atomicAdd(&s[S_DSQ_G1], v1*v1);
}

// center global views and cast to bf16
__global__ void center_cast_k(const float* g0, const float* g1,
                              const float* mean0, const float* mean1,
                              __bf16* o0, __bf16* o1) {
  size_t i = ((size_t)blockIdx.x * blockDim.x + threadIdx.x) * 4;
  const float* g = blockIdx.y ? g1 : g0;
  const float* m = blockIdx.y ? mean1 : mean0;
  __bf16* o = blockIdx.y ? o1 : o0;
  float4 v = *reinterpret_cast<const float4*>(g + i);
  size_t c = i & (GD - 1);
  o[i+0] = (__bf16)(v.x - m[c+0]);
  o[i+1] = (__bf16)(v.y - m[c+1]);
  o[i+2] = (__bf16)(v.z - m[c+2]);
  o[i+3] = (__bf16)(v.w - m[c+3]);
}

// plain f32 -> bf16 cast of the local feature maps
__global__ void cast_local_k(const float* z0, const float* z1,
                             __bf16* o0, __bf16* o1) {
  const float* src = blockIdx.y ? z1 : z0;
  __bf16* dst = blockIdx.y ? o1 : o0;
  size_t i = ((size_t)blockIdx.x * blockDim.x + threadIdx.x) * 4;
  float4 v = *reinterpret_cast<const float4*>(src + i);
  dst[i+0] = (__bf16)v.x; dst[i+1] = (__bf16)v.y;
  dst[i+2] = (__bf16)v.z; dst[i+3] = (__bf16)v.w;
}

// ||Xc Xcᵀ||_F² for the centered global views (256 x 8192)
__global__ void gram_fro_global_k(const __bf16* x0, const __bf16* x1, float* s) {
  int lane = threadIdx.x & 31, w = threadIdx.x >> 5;
  const __bf16* x = blockIdx.y ? x1 : x0;
  int slot = blockIdx.y ? S_FRO_G1 : S_FRO_G0;
  int tile = blockIdx.x * 4 + w;            // 64 tiles of 32x32 over 256x256
  int i0 = (tile >> 3) * 32, j0 = (tile & 7) * 32;
  float fro = gram_tile_fro32(x, GD, i0, j0, GD, lane, 1.f);
  if (lane == 0) atomicAdd(&s[slot], fro);
}

// Σcov² for cov = XcT·XcTᵀ/(n-1), XcT = centered-transposed matched sets
__global__ void gram_fro_local_k(const __bf16* fmatT, float* s) {
  int lane = threadIdx.x & 31, w = threadIdx.x >> 5;
  int m = blockIdx.y;                        // 8 matrices
  const __bf16* x = fmatT + (size_t)m * LC * LBM;
  int tile = blockIdx.x * 4 + w;             // 1024 tiles of 32x32 over 1024²
  int i0 = (tile >> 5) * 32, j0 = (tile & 31) * 32;
  float fro = gram_tile_fro32(x, LBM, i0, j0, LBM, lane, 1.f / (float)(LBM - 1));
  if (lane == 0) atomicAdd(&s[S_PAIR(m >> 1) + 5 + (m & 1)], fro);
}

// per-row squared L2 norms of the local feature maps
__global__ void rownorms_k(const float* z0, const float* z1,
                           float* n0, float* n1) {
  int r = blockIdx.x;
  const float* src = blockIdx.y ? z1 : z0;
  float* dst = blockIdx.y ? n1 : n0;
  const float* p = src + (size_t)r * LC;
  float acc = 0;
  for (int c = threadIdx.x; c < LC; c += 256) { float v = p[c]; acc += v*v; }
  acc = lane_sum32(acc);
  __shared__ float red[8];
  if ((threadIdx.x & 31) == 0) red[threadIdx.x >> 5] = acc;
  __syncthreads();
  if (threadIdx.x == 0) {
    float t = 0;
#pragma unroll
    for (int i = 0; i < 8; ++i) t += red[i];
    dst[r] = t;
  }
}

// batched feature cdist via WMMA (-2 A Bᵀ + norms), fused row-min/argmin.
// grid: (batch, 16 row-stripes); block: 256 = 8 waves; each wave owns a
// 16x32 column strip: one A fragment feeds two B fragments / accumulators.
__global__ void cdist_nn_k(const __bf16* za_all, const __bf16* zb_all,
                           const float* na_all, const float* nb_all,
                           float* md, int* mi) {
  int b = blockIdx.x, st = blockIdx.y;
  int lane = threadIdx.x & 31, w = threadIdx.x >> 5;
  const __bf16* za = za_all + (size_t)b * LN * LC;
  const __bf16* zb = zb_all + (size_t)b * LN * LC;
  const float* na = na_all + (size_t)b * LN;
  const float* nb = nb_all + (size_t)b * LN;
  int i0 = st * 16;
  int j0 = w * 32;
  v8f acc0 = {}, acc1 = {};
  for (int k = 0; k < LC; k += 32) {
    v16bf a  = frag_a_bf(za + (size_t)i0*LC + k, LC, lane);
    v16bf b0 = frag_b_bf(zb + (size_t)j0*LC + k, LC, lane);
    v16bf b1 = frag_b_bf(zb + (size_t)(j0+16)*LC + k, LC, lane);
    acc0 = wmma_bf16(a, b0, acc0);
    acc1 = wmma_bf16(a, b1, acc1);
  }
  float bestd[8]; int besti[8];
  int jA = j0 + (lane & 15), jB = jA + 16;
  float nbA = nb[jA], nbB = nb[jB];
#pragma unroll
  for (int v = 0; v < 8; ++v) {
    int i = i0 + (lane >> 4) * 8 + v;
    float ni = na[i];
    float dA = sqrtf(fmaxf(ni + nbA - 2.f * acc0[v], 0.f));
    float dB = sqrtf(fmaxf(ni + nbB - 2.f * acc1[v], 0.f));
    if (dA <= dB) { bestd[v] = dA; besti[v] = jA; }
    else          { bestd[v] = dB; besti[v] = jB; }
  }
  // reduce min(+argmin) over the 16 lanes of each half-wave
#pragma unroll
  for (int v = 0; v < 8; ++v) {
    float d = bestd[v]; int idx = besti[v];
#pragma unroll
    for (int m = 1; m < 16; m <<= 1) {
      float od = __shfl_xor(d, m); int oi = __shfl_xor(idx, m);
      if (od < d || (od == d && oi < idx)) { d = od; idx = oi; }
    }
    bestd[v] = d; besti[v] = idx;
  }
  __shared__ float sd[8][16];
  __shared__ int   si[8][16];
  if ((lane & 15) == 0) {
    int g = lane >> 4;
#pragma unroll
    for (int v = 0; v < 8; ++v) { sd[w][g*8+v] = bestd[v]; si[w][g*8+v] = besti[v]; }
  }
  __syncthreads();
  if (threadIdx.x < 16) {
    float d = sd[0][threadIdx.x]; int idx = si[0][threadIdx.x];
#pragma unroll
    for (int ww = 1; ww < 8; ++ww)
      if (sd[ww][threadIdx.x] < d) { d = sd[ww][threadIdx.x]; idx = si[ww][threadIdx.x]; }
    md[(size_t)b*LN + i0 + threadIdx.x] = d;
    mi[(size_t)b*LN + i0 + threadIdx.x] = idx;
  }
}

// grid-coordinate NN (C=2): one block per batch, thread per input position.
__global__ void grid_nn_k(const float* ga_all, const float* gb_all,
                          float* md, int* mi) {
  int b = blockIdx.x, i = threadIdx.x;
  __shared__ float gx[LN], gy[LN];
  const float* gb = gb_all + (size_t)b * LN * 2;
  gx[i] = gb[i*2]; gy[i] = gb[i*2+1];
  __syncthreads();
  const float* ga = ga_all + (size_t)b * LN * 2;
  float ax = ga[i*2], ay = ga[i*2+1];
  float best = 1e30f; int bi = 0;
  for (int j = 0; j < LN; ++j) {
    float dx = ax - gx[j], dy = ay - gy[j];
    float d2 = dx*dx + dy*dy;
    if (d2 < best) { best = d2; bi = j; }
  }
  md[(size_t)b*LN + i] = sqrtf(best);
  mi[(size_t)b*LN + i] = bi;
}

// select the NM smallest NN distances, keep position order; one wave/block.
__global__ void topk_k(const float* md_all, const int* mi_all,
                       int* sel, int* cand) {
  int b = blockIdx.x, a = blockIdx.y, lane = threadIdx.x;
  const float* md = md_all + ((size_t)a*LB + b) * LN;
  const int*   mi = mi_all + ((size_t)a*LB + b) * LN;
  float v[8];
#pragma unroll
  for (int t = 0; t < 8; ++t) v[t] = md[lane*8 + t];
  int picked[NM];
  for (int p = 0; p < NM; ++p) {
    float d = 1e30f; int pos = 0;
#pragma unroll
    for (int t = 0; t < 8; ++t)
      if (v[t] < d) { d = v[t]; pos = lane*8 + t; }
#pragma unroll
    for (int m = 1; m < 32; m <<= 1) {
      float od = __shfl_xor(d, m); int op = __shfl_xor(pos, m);
      if (od < d || (od == d && op < pos)) { d = od; pos = op; }
    }
    picked[p] = pos;
    if ((pos >> 3) == lane) v[pos & 7] = 1e30f;
  }
  if (lane == 0) {
    for (int x = 1; x < NM; ++x) {            // sort ascending (position order)
      int key = picked[x], y = x - 1;
      while (y >= 0 && picked[y] > key) { picked[y+1] = picked[y]; --y; }
      picked[y+1] = key;
    }
    for (int p = 0; p < NM; ++p) {
      sel [((size_t)a*LB + b)*NM + p] = picked[p];
      cand[((size_t)a*LB + b)*NM + p] = mi[picked[p]];
    }
  }
}

// gather matched bf16 rows into 8 matrices (LBM x LC); pure uint4 row copy
__global__ void gather_bf_k(const __bf16* l0, const __bf16* l1,
                            const int* sel, const int* cand, __bf16* fmat) {
  int m = blockIdx.y, r = blockIdx.x;
  int p = m >> 1, side = m & 1;
  bool swp = (p == 1) || (p == 3);               // dir1 pairs use (z1, z0)
  const __bf16* za = swp ? l1 : l0;
  const __bf16* zb = swp ? l0 : l1;
  const __bf16* src_base = side ? zb : za;
  const int*    idxarr   = side ? cand : sel;
  int b = r / NM, slot = r % NM;
  int row = idxarr[((size_t)p*LB + b)*NM + slot];
  const uint4* src = reinterpret_cast<const uint4*>(src_base + ((size_t)b*LN + row) * LC);
  uint4* dst = reinterpret_cast<uint4*>(fmat + ((size_t)m*LBM + r) * LC);
  for (int c = threadIdx.x; c < LC/8; c += blockDim.x) dst[c] = src[c];
}

// column stats of each matched pair: inv term, var terms, diag(cov)², means.
__global__ void local_colstats_k(const __bf16* fmat, float* colmean, float* s) {
  int p = blockIdx.y;
  int c = blockIdx.x * 256 + threadIdx.x;
  const __bf16* X = fmat + (size_t)(2*p)   * LBM * LC;
  const __bf16* Y = fmat + (size_t)(2*p+1) * LBM * LC;
  float sx = 0, qx = 0, sy = 0, qy = 0, inv = 0;
  for (int r = 0; r < LBM; ++r) {
    float x = (float)X[(size_t)r*LC + c];
    float y = (float)Y[(size_t)r*LC + c];
    sx += x; qx += x*x; sy += y; qy += y*y;
    float d = x - y; inv += d*d;
  }
  float mx = sx / LBM, my = sy / LBM;
  colmean[(size_t)(2*p)*LC + c]   = mx;
  colmean[(size_t)(2*p+1)*LC + c] = my;
  float vx = (qx - LBM*mx*mx) / (LBM - 1);
  float vy = (qy - LBM*my*my) / (LBM - 1);
  float* sp = s + S_PAIR(p);
  atomicAdd(&sp[0], inv);
  atomicAdd(&sp[1], fmaxf(0.f, 1.f - sqrtf(vx + EPS_)));
  atomicAdd(&sp[2], fmaxf(0.f, 1.f - sqrtf(vy + EPS_)));
  atomicAdd(&sp[3], vx*vx);
  atomicAdd(&sp[4], vy*vy);
}

// center + transpose: fmatT[m][c][r] = fmat[m][r][c] - mean[c]  (LDS tiled)
__global__ void center_transpose_k(const __bf16* fmat, const float* colmean,
                                   __bf16* fmatT) {
  int m = blockIdx.z;
  const __bf16* X = fmat + (size_t)m * LBM * LC;
  const float* mean = colmean + (size_t)m * LC;
  __bf16* XT = fmatT + (size_t)m * (size_t)LC * LBM;
  __shared__ __bf16 tile[32][33];
  int c0 = blockIdx.x * 32, r0 = blockIdx.y * 32;
  for (int rr = threadIdx.y; rr < 32; rr += 8) {
    int c = c0 + threadIdx.x;
    float v = (float)X[(size_t)(r0 + rr) * LC + c] - mean[c];
    tile[rr][threadIdx.x] = (__bf16)v;
  }
  __syncthreads();
  for (int cc = threadIdx.y; cc < 32; cc += 8)
    XT[(size_t)(c0 + cc) * LBM + r0 + threadIdx.x] = tile[threadIdx.x][cc];
}

__global__ void combine_k(const float* s, float* out) {
  // global term (symmetric: both orderings identical)
  float n = GB, d = GD;
  float inv = s[S_INV_G] / (n * d);
  float var = 0.5f * (s[S_VAR_G0] / d + s[S_VAR_G1] / d);
  float c0 = (s[S_FRO_G0] / ((n-1)*(n-1)) - s[S_DSQ_G0]) / d;
  float c1 = (s[S_FRO_G1] / ((n-1)*(n-1)) - s[S_DSQ_G1]) / d;
  float g = LAMBDA_*inv + MU_*var + NU_*(c0 + c1);
  // local term (the two _local_loss calls are identical)
  float vic[4];
  for (int p = 0; p < 4; ++p) {
    const float* sp = s + S_PAIR(p);
    float nl = LBM, dl = LC;
    float invp = sp[0] / (nl * dl);
    float varp = 0.5f * (sp[1] / dl + sp[2] / dl);
    float covx = (sp[5] - sp[3]) / dl;
    float covy = (sp[6] - sp[4]) / dl;
    vic[p] = LAMBDA_*invp + MU_*varp + NU_*(covx + covy);
  }
  float l = 0.5f*(vic[0] + vic[1]) + 0.5f*(vic[2] + vic[3]);
  out[0] = ALPHA_*g + (1.f - ALPHA_)*l;
}

// ---------------------------------------------------------------------------
extern "C" void kernel_launch(void* const* d_in, const int* in_sizes, int n_in,
                              void* d_out, int out_size, void* d_ws, size_t ws_size,
                              hipStream_t stream) {
  const float* g0   = (const float*)d_in[0];
  const float* g1   = (const float*)d_in[1];
  const float* zl0  = (const float*)d_in[2];
  const float* zl1  = (const float*)d_in[3];
  const float* gr0  = (const float*)d_in[4];
  const float* gr1  = (const float*)d_in[5];
  float* out = (float*)d_out;

  // workspace layout
  char* base = (char*)d_ws;
  size_t off = 0;
  float* s = (float*)(base + off);              off += 1024;
  float* mean_g0 = (float*)(base + off);        off += (size_t)GD * 4;
  float* mean_g1 = (float*)(base + off);        off += (size_t)GD * 4;
  __bf16* gbf0 = (__bf16*)(base + off);         off += (size_t)GB * GD * 2;
  __bf16* gbf1 = (__bf16*)(base + off);         off += (size_t)GB * GD * 2;
  __bf16* lbf0 = (__bf16*)(base + off);         off += (size_t)LB * LN * LC * 2;
  __bf16* lbf1 = (__bf16*)(base + off);         off += (size_t)LB * LN * LC * 2;
  float* norms0 = (float*)(base + off);         off += (size_t)LB * LN * 4;
  float* norms1 = (float*)(base + off);         off += (size_t)LB * LN * 4;
  float* md_all = (float*)(base + off);         off += (size_t)4 * LB * LN * 4;
  int*   mi_all = (int*)(base + off);           off += (size_t)4 * LB * LN * 4;
  int*   sel    = (int*)(base + off);           off += (size_t)4 * LB * NM * 4;
  int*   cand   = (int*)(base + off);           off += (size_t)4 * LB * NM * 4;
  float* colmean = (float*)(base + off);        off += (size_t)8 * LC * 4;
  __bf16* fmat  = (__bf16*)(base + off);        off += (size_t)8 * LBM * LC * 2;
  __bf16* fmatT = (__bf16*)(base + off);        off += (size_t)8 * LBM * LC * 2;

  init_scalars_k<<<1, 64, 0, stream>>>(s);

  // global branch
  global_colstats_k<<<GD/256, 256, 0, stream>>>(g0, g1, mean_g0, mean_g1, s);
  center_cast_k<<<dim3((GB*GD)/(256*4), 2), 256, 0, stream>>>(g0, g1, mean_g0, mean_g1, gbf0, gbf1);
  gram_fro_global_k<<<dim3(16, 2), 128, 0, stream>>>(gbf0, gbf1, s);

  // local branch: bf16 cast + NN matching
  cast_local_k<<<dim3(((size_t)LB*LN*LC)/(256*4), 2), 256, 0, stream>>>(zl0, zl1, lbf0, lbf1);
  rownorms_k<<<dim3(LB*LN, 2), 256, 0, stream>>>(zl0, zl1, norms0, norms1);
  cdist_nn_k<<<dim3(LB, 16), 256, 0, stream>>>(lbf0, lbf1, norms0, norms1,
                                               md_all + (size_t)0*LB*LN, mi_all + (size_t)0*LB*LN);
  cdist_nn_k<<<dim3(LB, 16), 256, 0, stream>>>(lbf1, lbf0, norms1, norms0,
                                               md_all + (size_t)1*LB*LN, mi_all + (size_t)1*LB*LN);
  grid_nn_k<<<LB, LN, 0, stream>>>(gr0, gr1, md_all + (size_t)2*LB*LN, mi_all + (size_t)2*LB*LN);
  grid_nn_k<<<LB, LN, 0, stream>>>(gr1, gr0, md_all + (size_t)3*LB*LN, mi_all + (size_t)3*LB*LN);
  topk_k<<<dim3(LB, 4), 32, 0, stream>>>(md_all, mi_all, sel, cand);

  // matched-set VICReg
  gather_bf_k<<<dim3(LBM, 8), 128, 0, stream>>>(lbf0, lbf1, sel, cand, fmat);
  local_colstats_k<<<dim3(LC/256, 4), 256, 0, stream>>>(fmat, colmean, s);
  center_transpose_k<<<dim3(LC/32, LBM/32, 8), dim3(32, 8), 0, stream>>>(fmat, colmean, fmatT);
  gram_fro_local_k<<<dim3(256, 8), 128, 0, stream>>>(fmatT, s);

  combine_k<<<1, 1, 0, stream>>>(s, out);
  (void)in_sizes; (void)n_in; (void)out_size; (void)ws_size;
}
